// ColBERT_10204842295904
// MI455X (gfx1250) — matrix operand order repaired and checked by reading, not compiled, not run on hardware
//
#include <hip/hip_runtime.h>

// ---------------------------------------------------------------------------
// ColBERT scoring for MI455X (gfx1250, wave32, WMMA).
// Roofline: D_emb stream = 402 MB @ 23.3 TB/s ~= 17 us floor; 27.7 GFLOP total.
// bf16 WMMA (16x16x32, fp32 accum) keeps compute (~5 us) under the HBM floor;
// fp32 WMMA (K=4) would be compute-bound, so operands are converted to bf16.
// ---------------------------------------------------------------------------

typedef __attribute__((ext_vector_type(16))) __bf16       v16bf;
typedef __attribute__((ext_vector_type(2)))  __bf16       v2bf;
typedef __attribute__((ext_vector_type(8)))  float        v8f;
typedef __attribute__((ext_vector_type(8)))  unsigned int u32x8;

// Native bf16 convert (backend emits v_cvt_pk_bf16_f32 when available; RNE).
__device__ __forceinline__ unsigned int pack2(float a, float b) {
    v2bf h = { (__bf16)a, (__bf16)b };
    return __builtin_bit_cast(unsigned int, h);
}
__device__ __forceinline__ v16bf mkfrag(uint4 a, uint4 b) {
    u32x8 u = {a.x, a.y, a.z, a.w, b.x, b.y, b.z, b.w};
    return __builtin_bit_cast(v16bf, u);
}

// ---------------------------------------------------------------------------
// Projection + L2 normalize:  Y[r, 0..127] = normalize(mask(r) * X[r,:] @ W^T)
// X: [rows,768] f32,  W: [128,768] f32,  Yn: [rows,128] bf16, ysq: [rows] f32
// Block = 256 threads (8 waves). Wave w owns output columns [16w,16w+16).
// W fragments (24 K-steps) persist in VGPRs; 16-row input tile staged in LDS.
// ---------------------------------------------------------------------------
__global__ __launch_bounds__(256, 1)
void proj_norm_kernel(const float* __restrict__ X, const float* __restrict__ W,
                      const int* __restrict__ ids,           // null => no mask
                      unsigned short* __restrict__ Yn, float* __restrict__ ysq,
                      int nTiles)
{
    // Row stride 392 dwords (768 bf16 + 32B pad) -> conflict-free ds_load_b128.
    __shared__ __align__(16) unsigned int sA[16 * 392];   // 25 KB bf16 A tile
    __shared__ __align__(16) float sOut[16 * 144];        // 9 KB f32 out tile (pad 144)
    __shared__ float sScale[16];

    const int t    = threadIdx.x;
    const int w    = t >> 5;
    const int lane = t & 31;
    const int half = lane >> 4;     // 0: lanes 0-15, 1: lanes 16-31
    const int m16  = lane & 15;

    // --- persistent B fragments: B[k][n] = W[dcol(n)][k], bf16 -----------------
    // B K order used here (contiguous): lane half 0 holds K=base..base+15, half 1 K=base+16..+31.
    const int dcol = w * 16 + m16;
    v16bf bw[24];
#pragma unroll
    for (int kb = 0; kb < 24; ++kb) {
        const float* wp = W + (size_t)dcol * 768 + kb * 32 + half * 16;
        float4 f0 = *(const float4*)(wp + 0);
        float4 f1 = *(const float4*)(wp + 4);
        float4 f2 = *(const float4*)(wp + 8);
        float4 f3 = *(const float4*)(wp + 12);
        u32x8 u = { pack2(f0.x, f0.y), pack2(f0.z, f0.w),
                    pack2(f1.x, f1.y), pack2(f1.z, f1.w),
                    pack2(f2.x, f2.y), pack2(f2.z, f2.w),
                    pack2(f3.x, f3.y), pack2(f3.z, f3.w) };
        bw[kb] = __builtin_bit_cast(v16bf, u);
    }

    for (int tile = blockIdx.x; tile < nTiles; tile += gridDim.x) {
        const long rowbase = (long)tile * 16;

        // --- stage A: 16x768 f32 -> bf16 in LDS (coalesced float4 stream) -----
#pragma unroll
        for (int i = 0; i < 12; ++i) {
            int q   = i * 256 + t;          // quad id, 3072 quads total
            int row = q / 192;
            int qc  = q - row * 192;
            float4 f = *(const float4*)(X + (rowbase + row) * 768 + qc * 4);
            *(uint2*)&sA[row * 392 + qc * 2] =
                make_uint2(pack2(f.x, f.y), pack2(f.z, f.w));
        }
        // prefetch next tile (global_prefetch_b8): 49 KB = 384 lines; 2 lines/thread.
        // Hides HBM latency behind the WMMA phase at zero VGPR cost.
        if (tile + gridDim.x < nTiles) {
            const float* nx = X + (rowbase + (long)gridDim.x * 16) * 768;
            __builtin_prefetch(nx + t * 48, 0, 1);
            __builtin_prefetch(nx + t * 48 + 32, 0, 1);
        }
        __syncthreads();

        // --- GEMM: 24 x v_wmma_f32_16x16x32_bf16 ------------------------------
        // A K order used here (contiguous): half 0: K = 0..15; half 1: K = 16..31 (matches B).
        v8f acc = {};
#pragma unroll
        for (int kb = 0; kb < 24; ++kb) {
            int di = m16 * 392 + kb * 16 + half * 8;       // dword index
            uint4 a0 = *(const uint4*)&sA[di];             // k .. k+7
            uint4 a1 = *(const uint4*)&sA[di + 4];         // k+8 .. k+15
            acc = __builtin_amdgcn_wmma_f32_16x16x32_bf16(
                      false, mkfrag(a0, a1), false, bw[kb],
                      (short)0, acc, false, false);
        }

        // --- C layout: m = r + 8*half, n = lane&15 ----------------------------
        {
            int c = w * 16 + m16;
#pragma unroll
            for (int r = 0; r < 8; ++r)
                sOut[(half * 8 + r) * 144 + c] = acc[r];
        }
        __syncthreads();

        // --- per-row norm / mask ---------------------------------------------
        if (t < 16) {
            float s = 0.f;
#pragma unroll
            for (int c = 0; c < 128; c += 4) {
                float4 v = *(const float4*)&sOut[t * 144 + c];
                s += v.x * v.x + v.y * v.y + v.z * v.z + v.w * v.w;
            }
            bool keep = true;
            if (ids) keep = (ids[rowbase + t] != 0);
            if (!keep) s = 0.f;
            float n  = sqrtf(s);
            float sc = keep ? (1.f / fmaxf(n, 1e-12f)) : 0.f;
            ysq[rowbase + t] = s * sc * sc;                // == |normalized row|^2
            sScale[t] = sc;
        }
        __syncthreads();

        // --- scale + bf16 store (row-major, 16B/lane coalesced) ---------------
        {
            int row = t >> 4;
            int c   = (t & 15) * 8;
            float sc = sScale[row];
            const float* p = &sOut[row * 144 + c];
            uint4 o;
            o.x = pack2(p[0] * sc, p[1] * sc);
            o.y = pack2(p[2] * sc, p[3] * sc);
            o.z = pack2(p[4] * sc, p[5] * sc);
            o.w = pack2(p[6] * sc, p[7] * sc);
            *(uint4*)(Yn + (rowbase + row) * 128 + c) = o;
        }
        // next iteration's sA writes are fenced by the post-GEMM __syncthreads
    }
}

// ---------------------------------------------------------------------------
// Score: per batch b, S[q,t] = 2*(Qn.Dn) - q_sq[q] - d_sq[t]; attn==0 -> NEG;
// out[b] = sum_q max_t S.   One block per batch, 8 waves over 64 token tiles.
// ---------------------------------------------------------------------------
__global__ __launch_bounds__(256, 1)
void score_kernel(const unsigned short* __restrict__ Qn,
                  const unsigned short* __restrict__ Dn,
                  const float* __restrict__ qsq, const float* __restrict__ dsq,
                  const int* __restrict__ attn, float* __restrict__ out)
{
    __shared__ float s_qsq[32];
    __shared__ float s_wmax[8][32];
    __shared__ int   s_any;

    const int b    = blockIdx.x;
    const int t    = threadIdx.x;
    const int w    = t >> 5;
    const int lane = t & 31;
    const int half = lane >> 4;
    const int m16  = lane & 15;
    const int aoff = half * 16;     // A fragment contiguous K
    const int boff = half * 16;     // B fragment contiguous K

    if (t == 0)  s_any = 0;
    if (t < 32)  s_qsq[t] = qsq[b * 32 + t];
    __syncthreads();
    {
        int any = 0;
#pragma unroll
        for (int i = 0; i < 4; ++i)
            if (attn[b * 1024 + t * 4 + i] == 0) any = 1;
        if (any) atomicOr(&s_any, 1);
    }

    // preload Q fragments: 2 M-tiles x 4 K-steps
    v16bf aq[2][4];
#pragma unroll
    for (int mt = 0; mt < 2; ++mt)
#pragma unroll
        for (int kb = 0; kb < 4; ++kb) {
            const unsigned short* p =
                Qn + ((long)b * 32 + mt * 16 + m16) * 128 + kb * 32 + aoff;
            aq[mt][kb] = mkfrag(*(const uint4*)p, *(const uint4*)(p + 8));
        }

    float runmax[2][8];
#pragma unroll
    for (int mt = 0; mt < 2; ++mt)
#pragma unroll
        for (int r = 0; r < 8; ++r) runmax[mt][r] = -3.0e30f;

    for (int nt = w; nt < 64; nt += 8) {               // 8 token-tiles per wave
        const int  tcol = nt * 16 + m16;
        const long gt   = (long)b * 1024 + tcol;
        const float bias = attn[gt] ? dsq[gt] : 1.0e30f;   // masked -> never wins
        v16bf bd[4];
#pragma unroll
        for (int kb = 0; kb < 4; ++kb) {
            const unsigned short* p = Dn + gt * 128 + kb * 32 + boff;
            bd[kb] = mkfrag(*(const uint4*)p, *(const uint4*)(p + 8));
        }
#pragma unroll
        for (int mt = 0; mt < 2; ++mt) {
            v8f acc = {};
#pragma unroll
            for (int kb = 0; kb < 4; ++kb)
                acc = __builtin_amdgcn_wmma_f32_16x16x32_bf16(
                          false, aq[mt][kb], false, bd[kb],
                          (short)0, acc, false, false);
#pragma unroll
            for (int r = 0; r < 8; ++r)                 // column n == this lane's tcol
                runmax[mt][r] = fmaxf(runmax[mt][r], 2.f * acc[r] - bias);
        }
    }

    // max over the 16 token-columns held across each 16-lane half
#pragma unroll
    for (int mt = 0; mt < 2; ++mt)
#pragma unroll
        for (int r = 0; r < 8; ++r) {
            float v = runmax[mt][r];
            v = fmaxf(v, __shfl_xor(v, 1, 32));
            v = fmaxf(v, __shfl_xor(v, 2, 32));
            v = fmaxf(v, __shfl_xor(v, 4, 32));
            v = fmaxf(v, __shfl_xor(v, 8, 32));
            if (m16 == 0) s_wmax[w][mt * 16 + r + half * 8] = v;
        }
    __syncthreads();

    if (t < 32) {                                      // wave 0 finishes up
        float mx = s_wmax[0][t];
#pragma unroll
        for (int ww = 1; ww < 8; ++ww) mx = fmaxf(mx, s_wmax[ww][t]);
        float fin = mx - s_qsq[t];
        if (s_any) fin = fmaxf(fin, -100000.0f);       // NEG from masked tokens
        fin += __shfl_xor(fin, 16, 32);
        fin += __shfl_xor(fin, 8, 32);
        fin += __shfl_xor(fin, 4, 32);
        fin += __shfl_xor(fin, 2, 32);
        fin += __shfl_xor(fin, 1, 32);
        if (t == 0) out[b] = fin;
    }
}

// ---------------------------------------------------------------------------
extern "C" void kernel_launch(void* const* d_in, const int* in_sizes, int n_in,
                              void* d_out, int out_size, void* d_ws, size_t ws_size,
                              hipStream_t stream)
{
    const float* Q_emb = (const float*)d_in[0];   // [128,32,768]
    const float* D_emb = (const float*)d_in[1];   // [128,1024,768]
    const int*   ids   = (const int*)d_in[2];     // [128,1024]
    const int*   attn  = (const int*)d_in[3];     // [128,1024]
    const float* W     = (const float*)d_in[4];   // [128,768]
    float*       out   = (float*)d_out;           // [128]

    // workspace layout (~35.1 MB)
    char* ws = (char*)d_ws;
    unsigned short* Dn  = (unsigned short*)(ws);                       // 33554432 B
    unsigned short* Qn  = (unsigned short*)(ws + 33554432);            //  1048576 B
    float*          dsq = (float*)(ws + 33554432 + 1048576);           //   524288 B
    float*          qsq = (float*)(ws + 33554432 + 1048576 + 524288);  //    16384 B

    proj_norm_kernel<<<512, 256, 0, stream>>>(D_emb, W, ids,     Dn, dsq, 8192);
    proj_norm_kernel<<<256, 256, 0, stream>>>(Q_emb, W, nullptr, Qn, qsq, 256);
    score_kernel<<<128, 256, 0, stream>>>(Qn, Dn, qsq, dsq, attn, out);
}
